// MultilayerPCN_28149215657992
// MI455X (gfx1250) — matrix operand order repaired and checked by reading, not compiled, not run on hardware
//
#include <hip/hip_runtime.h>
#include <hip/hip_bf16.h>
#include <math.h>

// ---------------------------------------------------------------------------
// MultilayerPCN on MI455X (gfx1250).
// bf16 WMMA GEMMs (v_wmma_f32_16x16x32_bf16), double-buffered async
// global->LDS tile staging (ASYNCcnt), fused PCN epilogues that also produce
// the next GEMM's bf16 A-operand.
// ---------------------------------------------------------------------------

typedef __bf16 bf16_t;
typedef __attribute__((ext_vector_type(16))) __bf16 v16bf;
typedef __attribute__((ext_vector_type(8)))  float  v8f;
typedef __attribute__((ext_vector_type(4)))  float  v4f;
typedef __attribute__((ext_vector_type(4)))  int    v4i_t;

#define AS1 __attribute__((address_space(1)))
#define AS3 __attribute__((address_space(3)))

#define PCN_N0 2048
#define PCN_N1 4096
#define PCN_N2 8192
#define PCN_B  256
#define PCN_LAMB 0.5f

#define BM 256            // full batch per block
#define BN 256
#define BK 32
#define LDA 40            // padded LDS row stride (bf16) = 80B (16B aligned)
#define LDB 40
#define NTHR 512          // 16 waves

enum { A_ID = 0, A_TANH = 1 };
enum { B_NT = 0, B_T = 1 };
enum { EPI_V0 = 0, EPI_V1 = 1, EPI_E1 = 2, EPI_E2 = 3 };

// ---- CDNA5 async global->LDS copy (ASYNCcnt path) -------------------------
__device__ __forceinline__ void async_copy_b128(const bf16_t* g, bf16_t* l)
{
#if defined(__HIP_DEVICE_COMPILE__)
#if __has_builtin(__builtin_amdgcn_global_load_async_to_lds_b128)
  __builtin_amdgcn_global_load_async_to_lds_b128(
      (AS1 v4i_t*)g, (AS3 v4i_t*)l, 0, 0);
#else
  unsigned loff = (unsigned)(unsigned long long)(AS3 char*)l;
  asm volatile("global_load_async_to_lds_b128 %0, %1, off"
               :: "v"(loff), "v"((unsigned long long)(size_t)g)
               : "memory");
#endif
#else
  (void)g; (void)l;
#endif
}

__device__ __forceinline__ void wait_async0()
{
#if defined(__HIP_DEVICE_COMPILE__)
#if __has_builtin(__builtin_amdgcn_s_wait_asynccnt)
  __builtin_amdgcn_s_wait_asynccnt(0);
#else
  asm volatile("s_wait_asynccnt 0x0" ::: "memory");
#endif
#endif
}

// Wait until only the 4 most recently issued async ops remain outstanding
// (i.e. the previous stage's copies have landed in LDS).
__device__ __forceinline__ void wait_async4()
{
#if defined(__HIP_DEVICE_COMPILE__)
#if __has_builtin(__builtin_amdgcn_s_wait_asynccnt)
  __builtin_amdgcn_s_wait_asynccnt(4);
#else
  asm volatile("s_wait_asynccnt 0x4" ::: "memory");
#endif
#endif
}

// ---------------------------------------------------------------------------
// GEMM: C[256, N] = op(A)[256, K] * B  (+ fused PCN epilogue)
//  PRE=1: A and B are pre-converted bf16 arrays in [row][K] layout ->
//         double-buffered async staging straight into LDS.
//  PRE=0: fallback, f32 sources converted (and optionally tanh'd) in staging.
//   B_NT: B(k,n) = Wf32[k*N + n]      B_T: B(k,n) = Wf32[n*K + k]
// ---------------------------------------------------------------------------
template<int PRE, int AMODE, int BMODE, int EPI>
__global__ __launch_bounds__(NTHR)
void pcn_gemm(const float* __restrict__ Af32, const bf16_t* __restrict__ Abf,
              const float* __restrict__ Wf32, const bf16_t* __restrict__ Wbf,
              int K, int N,
              const float* __restrict__ lr_p, const float* __restrict__ memv,
              float* __restrict__ v0, float* __restrict__ e0,
              float* __restrict__ v1, float* __restrict__ e1,
              float* __restrict__ e2, const float* __restrict__ binp,
              bf16_t* __restrict__ t0bf, bf16_t* __restrict__ t1bf,
              bf16_t* __restrict__ e1bf, bf16_t* __restrict__ e2bf,
              float* __restrict__ partial)
{
  __shared__ bf16_t sA[2 * BM * LDA];   // 2 buffers (PRE path); PRE=0 uses buf0
  __shared__ bf16_t sB[2 * BN * LDB];
  __shared__ float  red[NTHR];

  const int tid  = threadIdx.x;
  const int lane = tid & 31;
  const int wv   = tid >> 5;       // wave 0..15
  const int wm   = wv & 3;         // 64-row chunk
  const int wn   = wv >> 2;        // 64-col chunk
  const int half = lane >> 4;
  const int l15  = lane & 15;
  const int nBase = blockIdx.x * BN;

  v8f acc[4][4];
#pragma unroll
  for (int i = 0; i < 4; ++i)
#pragma unroll
    for (int j = 0; j < 4; ++j)
      acc[i][j] = (v8f)0.f;

  // async stage issue: 2x(A,B) b128 per thread into buffer `buf`
  auto stage_async = [&](int buf, int kk) {
    const int c  = tid & 3;        // 16B chunk within 64B row
    const int r0 = tid >> 2;       // 0..127
    bf16_t* sAb = sA + buf * (BM * LDA);
    bf16_t* sBb = sB + buf * (BN * LDB);
#pragma unroll
    for (int it = 0; it < 2; ++it) {
      const int row = r0 + it * 128;
      async_copy_b128(Abf + (size_t)row * K + kk + c * 8,
                      sAb + row * LDA + c * 8);
      async_copy_b128(Wbf + (size_t)(nBase + row) * K + kk + c * 8,
                      sBb + row * LDB + c * 8);
    }
  };

  if (PRE) stage_async(0, 0);      // prologue: tile 0 in flight

  int cur = 0;
  for (int k0 = 0; k0 < K; k0 += BK) {
    if (PRE) {
      // ---- double-buffered async staging ----
      const bool has_next = (k0 + BK) < K;
      if (has_next) stage_async(cur ^ 1, k0 + BK);   // overlap with this tile
      if (has_next) wait_async4();                   // tile `cur` landed
      else          wait_async0();
    } else {
      // ---- fallback staging: f32 -> bf16 (+ optional tanh on A) ----
      {
        const int c4 = tid & 7;     // float4 slot in 32-col row
        const int r0 = tid >> 3;    // 0..63
#pragma unroll
        for (int it = 0; it < 4; ++it) {
          const int row = r0 + it * 64;
          v4f a = *(const v4f*)(Af32 + (size_t)row * K + k0 + c4 * 4);
          if (AMODE == A_TANH) {
            a.x = tanhf(a.x); a.y = tanhf(a.y);
            a.z = tanhf(a.z); a.w = tanhf(a.w);
          }
          bf16_t* dst = sA + row * LDA + c4 * 4;
          dst[0] = (bf16_t)a.x; dst[1] = (bf16_t)a.y;
          dst[2] = (bf16_t)a.z; dst[3] = (bf16_t)a.w;
        }
      }
      if (BMODE == B_T) {
        const int c4  = tid & 7;
        const int nr0 = tid >> 3;   // 0..63
#pragma unroll
        for (int it = 0; it < 4; ++it) {
          const int nn = nr0 + it * 64;
          v4f b = *(const v4f*)(Wf32 + (size_t)(nBase + nn) * K + k0 + c4 * 4);
          bf16_t* dst = sB + nn * LDB + c4 * 4;
          dst[0] = (bf16_t)b.x; dst[1] = (bf16_t)b.y;
          dst[2] = (bf16_t)b.z; dst[3] = (bf16_t)b.w;
        }
      } else {
        const int n4  = tid & 63;   // float4 slot along 256 cols
        const int kk0 = tid >> 6;   // 0..7
#pragma unroll
        for (int it = 0; it < 4; ++it) {
          const int k = kk0 + it * 8;
          v4f b = *(const v4f*)(Wf32 + (size_t)(k0 + k) * N + nBase + n4 * 4);
          sB[(n4 * 4 + 0) * LDB + k] = (bf16_t)b.x;
          sB[(n4 * 4 + 1) * LDB + k] = (bf16_t)b.y;
          sB[(n4 * 4 + 2) * LDB + k] = (bf16_t)b.z;
          sB[(n4 * 4 + 3) * LDB + k] = (bf16_t)b.w;
        }
      }
    }
    __syncthreads();

    // ---- fragments: A and B both use contiguous k = 16*half + e ----
    const bf16_t* sAc = sA + (PRE ? cur * (BM * LDA) : 0);
    const bf16_t* sBc = sB + (PRE ? cur * (BN * LDB) : 0);
    v16bf afr[4], bfr[4];
#pragma unroll
    for (int i = 0; i < 4; ++i) {
      const bf16_t* ap = sAc + (wm * 64 + i * 16 + l15) * LDA;
#pragma unroll
      for (int e = 0; e < 16; ++e)
        afr[i][e] = ap[16 * half + e];
    }
#pragma unroll
    for (int j = 0; j < 4; ++j) {
      const bf16_t* bp = sBc + (wn * 64 + j * 16 + l15) * LDB + 16 * half;
#pragma unroll
      for (int e = 0; e < 16; ++e)
        bfr[j][e] = bp[e];
    }
#pragma unroll
    for (int i = 0; i < 4; ++i)
#pragma unroll
      for (int j = 0; j < 4; ++j)
        acc[i][j] = __builtin_amdgcn_wmma_f32_16x16x32_bf16(
            false, afr[i], false, bfr[j], (short)0, acc[i][j], false, false);
    __syncthreads();    // all waves done reading `cur` -> safe to rewrite
    cur ^= 1;
  }

  // ---- fused PCN epilogue (also emits next GEMM's bf16 A when PRE) ----
  float lsum = 0.f;
  const float lr = (EPI == EPI_V0 || EPI == EPI_V1) ? *lr_p : 0.f;
#pragma unroll
  for (int i = 0; i < 4; ++i)
#pragma unroll
    for (int j = 0; j < 4; ++j)
#pragma unroll
      for (int r = 0; r < 8; ++r) {
        const int m = wm * 64 + i * 16 + 8 * half + r;
        const int n = nBase + wn * 64 + j * 16 + l15;
        const size_t idx = (size_t)m * N + n;
        const float c = acc[i][j][r];
        if (EPI == EPI_V0) {
          const float v = v0[idx], e = e0[idx];
          const float t = tanhf(v);
          const float sg = (v > 0.f) ? 1.f : ((v < 0.f) ? -1.f : 0.f);
          const float d = -e - PCN_LAMB * sg + (1.f - t * t) * c;
          float nv = v + lr * d;
          nv = nv > 0.f ? nv : 0.f;
          const float ne = nv - memv[n];
          v0[idx] = nv; e0[idx] = ne;
          if (PRE) t0bf[idx] = (bf16_t)tanhf(nv);   // A of G3
          lsum += ne * ne;
        } else if (EPI == EPI_V1) {
          const float v = v1[idx], e = e1[idx];
          const float t = tanhf(v);
          const float d = -e + (1.f - t * t) * c;
          float nv = v + lr * d;
          nv = nv > 0.f ? nv : 0.f;
          v1[idx] = nv;
          if (PRE) t1bf[idx] = (bf16_t)tanhf(nv);   // A of G4
        } else if (EPI == EPI_E1) {
          const float ne = v1[idx] - c;
          e1[idx] = ne;
          if (PRE) e1bf[idx] = (bf16_t)ne;          // A of next G1
          lsum += ne * ne;
        } else {
          const float ne = binp[idx] - c;
          if (PRE) e2bf[idx] = (bf16_t)ne;          // A of next G2
          else     e2[idx]   = ne;
          lsum += ne * ne;
        }
      }

  if (EPI != EPI_V1) {
    red[tid] = lsum;
    __syncthreads();
    for (int s = NTHR / 2; s > 0; s >>= 1) {
      if (tid < s) red[tid] += red[tid + s];
      __syncthreads();
    }
    if (tid == 0) partial[blockIdx.x] = red[0];
  }
}

// ---------------------------------------------------------------------------
// One-time weight conversion: W[R][C] f32 -> Wc[R][C] bf16 and Wt[C][R] bf16.
// ---------------------------------------------------------------------------
__global__ __launch_bounds__(256)
void pcn_convert_w(const float* __restrict__ W, bf16_t* __restrict__ Wc,
                   bf16_t* __restrict__ Wt, int R, int C)
{
  __shared__ float tile[32][33];
  const int c0 = blockIdx.x * 32, r0 = blockIdx.y * 32;
  const int tx = threadIdx.x & 31, ty = threadIdx.x >> 5;   // 32x8
#pragma unroll
  for (int i = 0; i < 32; i += 8) {
    const float v = W[(size_t)(r0 + ty + i) * C + c0 + tx];
    tile[ty + i][tx] = v;
    Wc[(size_t)(r0 + ty + i) * C + c0 + tx] = (bf16_t)v;
  }
  __syncthreads();
#pragma unroll
  for (int i = 0; i < 32; i += 8)
    Wt[(size_t)(c0 + ty + i) * R + r0 + tx] = (bf16_t)tile[tx][ty + i];
}

// ---------------------------------------------------------------------------
// Init helpers
// ---------------------------------------------------------------------------
__global__ __launch_bounds__(256)
void pcn_gemv_tanh(const float* __restrict__ W, const float* __restrict__ x,
                   float* __restrict__ y, int K)
{
  __shared__ float red[256];
  const float* row = W + (size_t)blockIdx.x * K;
  float s = 0.f;
  for (int j = threadIdx.x; j < K; j += 256) s += row[j] * tanhf(x[j]);
  red[threadIdx.x] = s;
  __syncthreads();
  for (int st = 128; st > 0; st >>= 1) {
    if (threadIdx.x < st) red[threadIdx.x] += red[threadIdx.x + st];
    __syncthreads();
  }
  if (threadIdx.x == 0) y[blockIdx.x] = red[0];
}

__global__ __launch_bounds__(256)
void pcn_init_v0e0(const float* __restrict__ memv, float* __restrict__ v0,
                   float* __restrict__ e0)
{
  const int idx = blockIdx.x * 256 + threadIdx.x;
  v0[idx] = memv[idx & (PCN_N0 - 1)];
  e0[idx] = 0.f;
}

__global__ __launch_bounds__(256)
void pcn_init_v1e1(const float* __restrict__ v1row, float* __restrict__ v1,
                   float* __restrict__ e1)
{
  const int idx = blockIdx.x * 256 + threadIdx.x;
  v1[idx] = v1row[idx & (PCN_N1 - 1)];
  e1[idx] = 0.f;
}

__global__ __launch_bounds__(256)
void pcn_zero_bf16(bf16_t* __restrict__ p)
{
  p[blockIdx.x * 256 + threadIdx.x] = (bf16_t)0.f;
}

__global__ __launch_bounds__(256)
void pcn_init_e2_f32(const float* __restrict__ binp,
                     const float* __restrict__ p2row, float* __restrict__ e2)
{
  const int idx = blockIdx.x * 256 + threadIdx.x;
  e2[idx] = binp[idx] - p2row[idx & (PCN_N2 - 1)];
}

__global__ __launch_bounds__(256)
void pcn_init_e2_bf16(const float* __restrict__ binp,
                      const float* __restrict__ p2row, bf16_t* __restrict__ e2bf)
{
  const int idx = blockIdx.x * 256 + threadIdx.x;
  e2bf[idx] = (bf16_t)(binp[idx] - p2row[idx & (PCN_N2 - 1)]);
}

__global__ __launch_bounds__(256)
void pcn_reduce_loss(const float* __restrict__ partials, int n,
                     float* __restrict__ out_elem)
{
  __shared__ float red[256];
  float s = 0.f;
  for (int i = threadIdx.x; i < n; i += 256) s += partials[i];
  red[threadIdx.x] = s;
  __syncthreads();
  for (int st = 128; st > 0; st >>= 1) {
    if (threadIdx.x < st) red[threadIdx.x] += red[threadIdx.x + st];
    __syncthreads();
  }
  if (threadIdx.x == 0) *out_elem = red[0] * (100.0f / (float)PCN_B);
}

// ---------------------------------------------------------------------------
extern "C" void kernel_launch(void* const* d_in, const int* in_sizes, int n_in,
                              void* d_out, int out_size, void* d_ws, size_t ws_size,
                              hipStream_t stream)
{
  (void)in_sizes; (void)n_in;
  const float* binp = (const float*)d_in[0];   // [256, 8192]
  const float* W0   = (const float*)d_in[1];   // [4096, 2048]
  const float* W1   = (const float*)d_in[2];   // [8192, 4096]
  const float* memv = (const float*)d_in[3];   // [2048]
  const float* lr   = (const float*)d_in[5];   // inf_lr (device scalar)
  float* out = (float*)d_out;
  const int n_iters = out_size;

  // ---- workspace layout (f32 prefix shared by both paths) ----
  float* fp = (float*)d_ws;
  float* v0    = fp;                      fp += PCN_B * PCN_N0;
  float* e0    = fp;                      fp += PCN_B * PCN_N0;
  float* v1    = fp;                      fp += PCN_B * PCN_N1;
  float* e1    = fp;                      fp += PCN_B * PCN_N1;
  float* e2    = fp;                      fp += PCN_B * PCN_N2;  // fallback only
  float* v1row = fp;                      fp += PCN_N1;
  float* p2row = fp;                      fp += PCN_N2;
  float* parts = fp;                      fp += 64;
  bf16_t* bp = (bf16_t*)fp;
  bf16_t* t0bf = bp;                      bp += PCN_B * PCN_N0;
  bf16_t* t1bf = bp;                      bp += PCN_B * PCN_N1;
  bf16_t* e1bf = bp;                      bp += PCN_B * PCN_N1;
  bf16_t* e2bf = bp;                      bp += PCN_B * PCN_N2;
  bf16_t* w0c  = bp;                      bp += (size_t)PCN_N1 * PCN_N0;
  bf16_t* w0t  = bp;                      bp += (size_t)PCN_N0 * PCN_N1;
  bf16_t* w1c  = bp;                      bp += (size_t)PCN_N2 * PCN_N1;
  bf16_t* w1t  = bp;                      bp += (size_t)PCN_N1 * PCN_N2;
  const size_t fast_need = (size_t)((char*)bp - (char*)d_ws);
  const bool fast = ws_size >= fast_need;

  // ---- init: v0 = memory (bcast), e0 = e1 = 0,
  //      v1 row = W0 @ tanh(memory), e2 = batch_inp - W1 @ tanh(v1row) ----
  pcn_init_v0e0<<<PCN_B * PCN_N0 / 256, 256, 0, stream>>>(memv, v0, e0);
  pcn_gemv_tanh<<<PCN_N1, 256, 0, stream>>>(W0, memv, v1row, PCN_N0);
  pcn_init_v1e1<<<PCN_B * PCN_N1 / 256, 256, 0, stream>>>(v1row, v1, e1);
  pcn_gemv_tanh<<<PCN_N2, 256, 0, stream>>>(W1, v1row, p2row, PCN_N1);

  if (fast) {
    pcn_convert_w<<<dim3(PCN_N0 / 32, PCN_N1 / 32), 256, 0, stream>>>(
        W0, w0c, w0t, PCN_N1, PCN_N0);
    pcn_convert_w<<<dim3(PCN_N1 / 32, PCN_N2 / 32), 256, 0, stream>>>(
        W1, w1c, w1t, PCN_N2, PCN_N1);
    pcn_zero_bf16<<<PCN_B * PCN_N1 / 256, 256, 0, stream>>>(e1bf);
    pcn_init_e2_bf16<<<PCN_B * PCN_N2 / 256, 256, 0, stream>>>(binp, p2row, e2bf);

    for (int it = 0; it < n_iters; ++it) {
      // G1: e1 @ W0 -> v0/e0 update (+ t0bf = bf16(tanh(nv0)))
      pcn_gemm<1, A_ID, B_NT, EPI_V0><<<PCN_N0 / BN, NTHR, 0, stream>>>(
          nullptr, e1bf, nullptr, w0t, PCN_N1, PCN_N0, lr, memv,
          v0, e0, v1, e1, e2, binp, t0bf, t1bf, e1bf, e2bf, parts + 0);
      // G2: e2 @ W1 -> v1 update (+ t1bf = bf16(tanh(nv1)))
      pcn_gemm<1, A_ID, B_NT, EPI_V1><<<PCN_N1 / BN, NTHR, 0, stream>>>(
          nullptr, e2bf, nullptr, w1t, PCN_N2, PCN_N1, lr, memv,
          v0, e0, v1, e1, e2, binp, t0bf, t1bf, e1bf, e2bf, parts);
      // G3: ne1 = v1 - tanh(v0) @ W0.T   (A = t0bf)
      pcn_gemm<1, A_TANH, B_T, EPI_E1><<<PCN_N1 / BN, NTHR, 0, stream>>>(
          nullptr, t0bf, nullptr, w0c, PCN_N0, PCN_N1, lr, memv,
          v0, e0, v1, e1, e2, binp, t0bf, t1bf, e1bf, e2bf, parts + 8);
      // G4: ne2 = binp - tanh(v1) @ W1.T  (A = t1bf)
      pcn_gemm<1, A_TANH, B_T, EPI_E2><<<PCN_N2 / BN, NTHR, 0, stream>>>(
          nullptr, t1bf, nullptr, w1c, PCN_N1, PCN_N2, lr, memv,
          v0, e0, v1, e1, e2, binp, t0bf, t1bf, e1bf, e2bf, parts + 24);
      pcn_reduce_loss<<<1, 256, 0, stream>>>(parts, 56, out + it);
    }
  } else {
    // ---- fallback: f32 sources, convert in staging (small ws) ----
    pcn_init_e2_f32<<<PCN_B * PCN_N2 / 256, 256, 0, stream>>>(binp, p2row, e2);

    for (int it = 0; it < n_iters; ++it) {
      pcn_gemm<0, A_ID, B_NT, EPI_V0><<<PCN_N0 / BN, NTHR, 0, stream>>>(
          e1, nullptr, W0, nullptr, PCN_N1, PCN_N0, lr, memv,
          v0, e0, v1, e1, e2, binp, nullptr, nullptr, nullptr, nullptr, parts + 0);
      pcn_gemm<0, A_ID, B_NT, EPI_V1><<<PCN_N1 / BN, NTHR, 0, stream>>>(
          e2, nullptr, W1, nullptr, PCN_N2, PCN_N1, lr, memv,
          v0, e0, v1, e1, e2, binp, nullptr, nullptr, nullptr, nullptr, parts);
      pcn_gemm<0, A_TANH, B_T, EPI_E1><<<PCN_N1 / BN, NTHR, 0, stream>>>(
          v0, nullptr, W0, nullptr, PCN_N0, PCN_N1, lr, memv,
          v0, e0, v1, e1, e2, binp, nullptr, nullptr, nullptr, nullptr, parts + 8);
      pcn_gemm<0, A_TANH, B_T, EPI_E2><<<PCN_N2 / BN, NTHR, 0, stream>>>(
          v1, nullptr, W1, nullptr, PCN_N1, PCN_N2, lr, memv,
          v0, e0, v1, e1, e2, binp, nullptr, nullptr, nullptr, nullptr, parts + 24);
      pcn_reduce_loss<<<1, 256, 0, stream>>>(parts, 56, out + it);
    }
  }
}
